// SingleMambaBlock_45071386805022
// MI455X (gfx1250) — compile-verified, matrix operand order checked by reading, not run on hardware
//
#include <hip/hip_runtime.h>

// ---------------------------------------------------------------------------
// Bidirectional Mamba (Vim) block for MI455X / gfx1250, wave32.
//   B=2, C=64, H=W=128 -> L=16384, d_inner=128, d_state=16, dt_rank=4, k=4
// GEMMs use v_wmma_f32_16x16x32_f16; scan uses chunked parallel recurrence.
// Round 2: fast-rcp SiLU (kills v_div_* expansion), uniform epilogue branch.
// ---------------------------------------------------------------------------

typedef __attribute__((ext_vector_type(16))) _Float16 v16h;
typedef __attribute__((ext_vector_type(8)))  _Float16 v8h;
typedef __attribute__((ext_vector_type(8)))  float    v8f;
typedef __attribute__((ext_vector_type(4)))  float    f4;

#define BATCHN 2
#define CDIM   64
#define LSEQ   16384
#define BL     32768          // BATCHN * LSEQ
#define DI     128            // d_inner
#define NS     16             // d_state
#define XD     48             // x_dbl row stride (36 padded to 3*16)
#define NCH    128            // scan chunks
#define CS     128            // chunk size (NCH*CS == LSEQ)

// silu(x) = x * rcp(1 + exp(-x)) using hardware v_rcp_f32 / v_exp_f32
__device__ __forceinline__ float siluf(float x) {
    return x * __builtin_amdgcn_rcpf(1.0f + __expf(-x));
}
__device__ __forceinline__ float softplusf(float x) {
    return (x > 20.0f) ? x : __logf(1.0f + __expf(x));
}

// --- WMMA fragment loaders -------------------------------------------------
// CDNA5 16-bit A-matrix 16x32 layout: lane&15 = matrix row (A) / col (B),
// lane>>4 = K-half. Elements 0..7 = K[k0 .. k0+7], 8..15 = K[k0+16 .. k0+23]
// where k0 = kblock*32 + (lane>>4)*8.
__device__ __forceinline__ v16h load_frag_h(const _Float16* row, int kblock, int khalf) {
    const int k0 = kblock * 32 + khalf * 8;
    const v8h c0 = *(const v8h*)(row + k0);
    const v8h c1 = *(const v8h*)(row + k0 + 16);
    v16h a;
#pragma unroll
    for (int i = 0; i < 8; ++i) { a[i] = c0[i]; a[8 + i] = c1[i]; }
    return a;
}

__device__ __forceinline__ v16h load_frag_f32(const float* row, int kblock, int khalf) {
    const int k0 = kblock * 32 + khalf * 8;
    const f4 c0 = *(const f4*)(row + k0);
    const f4 c1 = *(const f4*)(row + k0 + 4);
    const f4 c2 = *(const f4*)(row + k0 + 16);
    const f4 c3 = *(const f4*)(row + k0 + 20);
    v16h a;
#pragma unroll
    for (int i = 0; i < 4; ++i) {
        a[i]      = (_Float16)c0[i];
        a[4 + i]  = (_Float16)c1[i];
        a[8 + i]  = (_Float16)c2[i];
        a[12 + i] = (_Float16)c3[i];
    }
    return a;
}

// --- 1. LayerNorm over channels (writes f16 for WMMA A-operand) ------------
__global__ __launch_bounds__(256) void k_ln(const float* __restrict__ in,
                                            const float* __restrict__ g,
                                            const float* __restrict__ bta,
                                            _Float16* __restrict__ seqln) {
    const int lane = threadIdx.x & 31;
    const int row  = blockIdx.x * (blockDim.x >> 5) + (threadIdx.x >> 5);
    const int b = row >> 14, l = row & (LSEQ - 1);
    const float* base = in + (size_t)b * CDIM * LSEQ + l;
    float v0 = base[(size_t)lane * LSEQ];
    float v1 = base[(size_t)(lane + 32) * LSEQ];
    float s = v0 + v1;
#pragma unroll
    for (int o = 16; o; o >>= 1) s += __shfl_xor(s, o);
    const float mu = s * (1.0f / 64.0f);
    const float d0 = v0 - mu, d1 = v1 - mu;
    float q = d0 * d0 + d1 * d1;
#pragma unroll
    for (int o = 16; o; o >>= 1) q += __shfl_xor(q, o);
    const float rstd = rsqrtf(q * (1.0f / 64.0f) + 1e-5f);
    _Float16* orow = seqln + (size_t)row * CDIM;
    orow[lane]      = (_Float16)(d0 * rstd * g[lane] + bta[lane]);
    orow[lane + 32] = (_Float16)(d1 * rstd * g[lane + 32] + bta[lane + 32]);
}

// --- 2. in_proj GEMM: (BL x 64) x (64 -> 256); epilogue splits x | silu(z) -
__global__ __launch_bounds__(128) void k_inproj(const _Float16* __restrict__ seqln,
                                                const float* __restrict__ W,
                                                float* __restrict__ xbuf,
                                                float* __restrict__ siluz) {
    const int lane  = threadIdx.x & 31;
    const int tid   = blockIdx.x * (blockDim.x >> 5) + (threadIdx.x >> 5);
    const int NT    = 16;                         // 256/16 column tiles
    const int mt    = tid / NT, nt = tid % NT;
    const int khalf = lane >> 4;
    const int mr    = lane & 15;
    const _Float16* arow = seqln + (size_t)(mt * 16 + mr) * CDIM;
    const float*    brow = W     + (size_t)(nt * 16 + mr) * CDIM;
    v8f acc = {};
#pragma unroll
    for (int kb = 0; kb < 2; ++kb) {
        v16h a = load_frag_h(arow, kb, khalf);
        v16h bm = load_frag_f32(brow, kb, khalf);
        acc = __builtin_amdgcn_wmma_f32_16x16x32_f16(false, a, false, bm,
                                                     (short)0, acc, false, false);
    }
    const int mbase = mt * 16 + (khalf ? 8 : 0);
    if (nt < 8) {                                 // wave-uniform: x half
        const int nout = nt * 16 + mr;
#pragma unroll
        for (int r = 0; r < 8; ++r)
            xbuf[(size_t)(mbase + r) * DI + nout] = acc[r];
    } else {                                      // wave-uniform: z half -> silu
        const int nout = (nt - 8) * 16 + mr;
#pragma unroll
        for (int r = 0; r < 8; ++r)
            siluz[(size_t)(mbase + r) * DI + nout] = siluf(acc[r]);
    }
}

// --- 3. causal depthwise conv (k=4) + SiLU; flip handled by indexing -------
__global__ __launch_bounds__(256) void k_conv(const float* __restrict__ x,
                                              const float* __restrict__ w,
                                              const float* __restrict__ bias,
                                              float* __restrict__ u, int flip) {
    const size_t gid = (size_t)blockIdx.x * blockDim.x + threadIdx.x;
    const int d = (int)(gid & (DI - 1));
    const int m = (int)(gid >> 7);
    const int b = m >> 14, l = m & (LSEQ - 1);
    float acc = bias[d];
#pragma unroll
    for (int j = 0; j < 4; ++j) {
        const int t = l - 3 + j;
        if (t >= 0) {
            const int src = flip ? (LSEQ - 1 - t) : t;
            acc += w[d * 4 + j] * x[((size_t)b * LSEQ + src) * DI + d];
        }
    }
    u[(size_t)m * DI + d] = siluf(acc);
}

// --- 4. x_proj GEMM: (BL x 128) x (128 -> 36, padded 48) -------------------
__global__ __launch_bounds__(128) void k_xproj(const float* __restrict__ u,
                                               const float* __restrict__ W,
                                               float* __restrict__ xdbl) {
    const int lane  = threadIdx.x & 31;
    const int tid   = blockIdx.x * (blockDim.x >> 5) + (threadIdx.x >> 5);
    const int NT    = 3;                          // 48/16 column tiles
    const int mt    = tid / NT, nt = tid % NT;
    const int khalf = lane >> 4;
    const int mr    = lane & 15;
    const int n     = nt * 16 + mr;
    const bool nvalid = (n < 36);
    const float* arow = u + (size_t)(mt * 16 + mr) * DI;
    const float* brow = W + (size_t)(nvalid ? n : 0) * DI;
    v8f acc = {};
#pragma unroll
    for (int kb = 0; kb < 4; ++kb) {
        v16h a = load_frag_f32(arow, kb, khalf);
        v16h bm;
        if (nvalid) bm = load_frag_f32(brow, kb, khalf);
        else { v16h z = {}; bm = z; }
        acc = __builtin_amdgcn_wmma_f32_16x16x32_f16(false, a, false, bm,
                                                     (short)0, acc, false, false);
    }
    const int mbase = mt * 16 + (khalf ? 8 : 0);
#pragma unroll
    for (int r = 0; r < 8; ++r)
        xdbl[(size_t)(mbase + r) * XD + n] = acc[r];
}

// --- 5. dt = softplus(dt_low @ dtproj_w^T + bias) --------------------------
__global__ __launch_bounds__(256) void k_dt(const float* __restrict__ xdbl,
                                            const float* __restrict__ W,
                                            const float* __restrict__ bias,
                                            float* __restrict__ dt) {
    const size_t gid = (size_t)blockIdx.x * blockDim.x + threadIdx.x;
    const int d = (int)(gid & (DI - 1));
    const int m = (int)(gid >> 7);
    const float* xr = xdbl + (size_t)m * XD;
    float acc = bias[d];
#pragma unroll
    for (int r = 0; r < 4; ++r) acc += xr[r] * W[d * 4 + r];
    dt[(size_t)m * DI + d] = softplusf(acc);
}

// --- 6. scan phase 1: per-chunk local state product (h0 = 0) ---------------
// wave32: lanes 0..15 -> (d0, n=lane), lanes 16..31 -> (d0+1, n=lane-16)
__global__ __launch_bounds__(256) void k_scan1(const float* __restrict__ dt,
                                               const float* __restrict__ u,
                                               const float* __restrict__ xdbl,
                                               const float* __restrict__ A_log,
                                               float* __restrict__ hfin,
                                               float* __restrict__ aprod) {
    const int lane = threadIdx.x & 31;
    const int dpair = blockIdx.x * (blockDim.x >> 5) + (threadIdx.x >> 5);
    const int d = dpair * 2 + (lane >> 4);
    const int n = lane & 15;
    const int b = blockIdx.z, c = blockIdx.y;
    const float A = -__expf(A_log[d * NS + n]);
    float h = 0.0f, ap = 1.0f;
    const int mbase = b * LSEQ + c * CS;
    for (int t = 0; t < CS; ++t) {
        const size_t m = (size_t)(mbase + t);
        if ((t & 7) == 0) {
            __builtin_prefetch(&dt[(m + 16) * DI + d], 0, 0);
            __builtin_prefetch(&xdbl[(m + 16) * XD + 4 + n], 0, 0);
        }
        const float dtv = dt[m * DI + d];
        const float uv  = u[m * DI + d];
        const float Bv  = xdbl[m * XD + 4 + n];
        const float dA  = __expf(dtv * A);
        h  = fmaf(dA, h, dtv * Bv * uv);
        ap *= dA;
    }
    const size_t idx = (((size_t)b * NCH + c) * DI + d) * NS + n;
    hfin[idx]  = h;
    aprod[idx] = ap;
}

// --- 7. sequential chunk-prefix combine (depth NCH=128, tiny) --------------
__global__ __launch_bounds__(256) void k_comb(const float* __restrict__ hfin,
                                              const float* __restrict__ aprod,
                                              float* __restrict__ h0) {
    const int gid = blockIdx.x * blockDim.x + threadIdx.x;   // BATCHN*DI*NS
    const int n = gid & 15, d = (gid >> 4) & (DI - 1), b = gid >> 11;
    float H = 0.0f;
    for (int c = 0; c < NCH; ++c) {
        const size_t idx = (((size_t)b * NCH + c) * DI + d) * NS + n;
        h0[idx] = H;
        H = fmaf(aprod[idx], H, hfin[idx]);
    }
}

// --- 8. scan phase 3: rescan with true h0, reduce over n, gate, (un)flip ---
__global__ __launch_bounds__(256) void k_scan3(const float* __restrict__ dt,
                                               const float* __restrict__ u,
                                               const float* __restrict__ xdbl,
                                               const float* __restrict__ A_log,
                                               const float* __restrict__ Dv,
                                               const float* __restrict__ h0buf,
                                               const float* __restrict__ siluz,
                                               float* __restrict__ ysum, int flip) {
    const int lane = threadIdx.x & 31;
    const int dpair = blockIdx.x * (blockDim.x >> 5) + (threadIdx.x >> 5);
    const int d = dpair * 2 + (lane >> 4);
    const int n = lane & 15;
    const int b = blockIdx.z, c = blockIdx.y;
    const float A  = -__expf(A_log[d * NS + n]);
    const float Dd = Dv[d];
    float h = h0buf[(((size_t)b * NCH + c) * DI + d) * NS + n];
    for (int t = 0; t < CS; ++t) {
        const int l = c * CS + t;
        const size_t m = (size_t)b * LSEQ + l;
        if ((t & 7) == 0) {
            __builtin_prefetch(&dt[(m + 16) * DI + d], 0, 0);
            __builtin_prefetch(&xdbl[(m + 16) * XD + 4 + n], 0, 0);
        }
        const float dtv = dt[m * DI + d];
        const float uv  = u[m * DI + d];
        const float Bv  = xdbl[m * XD + 4 + n];
        const float Cv  = xdbl[m * XD + 20 + n];
        const float dA  = __expf(dtv * A);
        h = fmaf(dA, h, dtv * Bv * uv);
        float yp = h * Cv;                         // reduce over 16 states
        yp += __shfl_xor(yp, 1);
        yp += __shfl_xor(yp, 2);
        yp += __shfl_xor(yp, 4);
        yp += __shfl_xor(yp, 8);
        if (n == 0) {
            const float y  = yp + uv * Dd;
            const int   lo = flip ? (LSEQ - 1 - l) : l;
            const size_t mo = (size_t)b * LSEQ + lo;
            const float g = siluz[mo * DI + d];    // gate == silu(z) at output pos
            if (flip) ysum[mo * DI + d] += y * g;  // stream-ordered after fwd pass
            else      ysum[mo * DI + d]  = y * g;
        }
    }
}

// --- 9. out_proj GEMM: (BL x 128) x (128 -> 64); epilogue adds skip, NCHW --
__global__ __launch_bounds__(128) void k_outproj(const float* __restrict__ ysum,
                                                 const float* __restrict__ W,
                                                 const float* __restrict__ in,
                                                 float* __restrict__ out) {
    const int lane  = threadIdx.x & 31;
    const int tid   = blockIdx.x * (blockDim.x >> 5) + (threadIdx.x >> 5);
    const int NT    = 4;                          // 64/16 column tiles
    const int mt    = tid / NT, nt = tid % NT;
    const int khalf = lane >> 4;
    const int mr    = lane & 15;
    const float* arow = ysum + (size_t)(mt * 16 + mr) * DI;
    const float* brow = W    + (size_t)(nt * 16 + mr) * DI;
    v8f acc = {};
#pragma unroll
    for (int kb = 0; kb < 4; ++kb) {
        v16h a  = load_frag_f32(arow, kb, khalf);
        v16h bm = load_frag_f32(brow, kb, khalf);
        acc = __builtin_amdgcn_wmma_f32_16x16x32_f16(false, a, false, bm,
                                                     (short)0, acc, false, false);
    }
    const int mbase = mt * 16 + (khalf ? 8 : 0);
    const int nout  = nt * 16 + mr;
#pragma unroll
    for (int r = 0; r < 8; ++r) {
        const int mm = mbase + r;
        const int b = mm >> 14, l = mm & (LSEQ - 1);
        const size_t addr = ((size_t)b * CDIM + nout) * LSEQ + l;
        out[addr] = acc[r] + in[addr];
    }
}

// ---------------------------------------------------------------------------
extern "C" void kernel_launch(void* const* d_in, const int* in_sizes, int n_in,
                              void* d_out, int out_size, void* d_ws, size_t ws_size,
                              hipStream_t stream) {
    const float* in        = (const float*)d_in[0];
    const float* ln_g      = (const float*)d_in[1];
    const float* ln_b      = (const float*)d_in[2];
    const float* in_proj_w = (const float*)d_in[3];
    const float* conv_w_f  = (const float*)d_in[4];
    const float* conv_b_f  = (const float*)d_in[5];
    const float* xproj_w_f = (const float*)d_in[6];
    const float* dtproj_w_f= (const float*)d_in[7];
    const float* dtproj_b_f= (const float*)d_in[8];
    const float* A_log_f   = (const float*)d_in[9];
    const float* D_f       = (const float*)d_in[10];
    const float* conv_w_b  = (const float*)d_in[11];
    const float* conv_b_b  = (const float*)d_in[12];
    const float* xproj_w_b = (const float*)d_in[13];
    const float* dtproj_w_b= (const float*)d_in[14];
    const float* dtproj_b_b= (const float*)d_in[15];
    const float* A_log_b   = (const float*)d_in[16];
    const float* D_b       = (const float*)d_in[17];
    const float* out_proj_w= (const float*)d_in[18];
    float* out = (float*)d_out;

    // Workspace layout (all L2-resident on MI455X's 192MB L2).
    char* w = (char*)d_ws;
    size_t off = 0;
    auto alloc = [&](size_t bytes) -> void* {
        void* p = w + off;
        off += (bytes + 255) & ~(size_t)255;
        return p;
    };
    _Float16* seqln = (_Float16*)alloc((size_t)BL * CDIM * 2);
    float* xbuf   = (float*)alloc((size_t)BL * DI * 4);
    float* siluz  = (float*)alloc((size_t)BL * DI * 4);
    float* u_f    = (float*)alloc((size_t)BL * DI * 4);
    float* u_b    = (float*)alloc((size_t)BL * DI * 4);
    float* xdbl_f = (float*)alloc((size_t)BL * XD * 4);
    float* xdbl_b = (float*)alloc((size_t)BL * XD * 4);
    float* dt_f   = (float*)alloc((size_t)BL * DI * 4);
    float* dt_b   = (float*)alloc((size_t)BL * DI * 4);
    const size_t stbytes = (size_t)BATCHN * NCH * DI * NS * 4;
    float* hfin_f = (float*)alloc(stbytes);
    float* hfin_b = (float*)alloc(stbytes);
    float* apr_f  = (float*)alloc(stbytes);
    float* apr_b  = (float*)alloc(stbytes);
    float* h0_f   = (float*)alloc(stbytes);
    float* h0_b   = (float*)alloc(stbytes);
    float* yb     = (float*)alloc((size_t)BL * DI * 4);

    // 1. LayerNorm
    k_ln<<<BL / 8, 256, 0, stream>>>(in, ln_g, ln_b, seqln);
    // 2. in_proj (WMMA), split x | silu(z)
    k_inproj<<<(BL / 16) * 16 / 4, 128, 0, stream>>>(seqln, in_proj_w, xbuf, siluz);
    // 3. causal conv + SiLU, both directions
    k_conv<<<(BL * DI) / 256, 256, 0, stream>>>(xbuf, conv_w_f, conv_b_f, u_f, 0);
    k_conv<<<(BL * DI) / 256, 256, 0, stream>>>(xbuf, conv_w_b, conv_b_b, u_b, 1);
    // 4. x_proj (WMMA), both branches
    k_xproj<<<(BL / 16) * 3 / 4, 128, 0, stream>>>(u_f, xproj_w_f, xdbl_f);
    k_xproj<<<(BL / 16) * 3 / 4, 128, 0, stream>>>(u_b, xproj_w_b, xdbl_b);
    // 5. dt projection + softplus
    k_dt<<<(BL * DI) / 256, 256, 0, stream>>>(xdbl_f, dtproj_w_f, dtproj_b_f, dt_f);
    k_dt<<<(BL * DI) / 256, 256, 0, stream>>>(xdbl_b, dtproj_w_b, dtproj_b_b, dt_b);
    // 6-8. chunked parallel selective scan
    dim3 sgrid(8, NCH, BATCHN);      // 8 blocks x 8 waves = 64 d-pairs
    k_scan1<<<sgrid, 256, 0, stream>>>(dt_f, u_f, xdbl_f, A_log_f, hfin_f, apr_f);
    k_scan1<<<sgrid, 256, 0, stream>>>(dt_b, u_b, xdbl_b, A_log_b, hfin_b, apr_b);
    k_comb<<<(BATCHN * DI * NS) / 256, 256, 0, stream>>>(hfin_f, apr_f, h0_f);
    k_comb<<<(BATCHN * DI * NS) / 256, 256, 0, stream>>>(hfin_b, apr_b, h0_b);
    k_scan3<<<sgrid, 256, 0, stream>>>(dt_f, u_f, xdbl_f, A_log_f, D_f, h0_f, siluz, yb, 0);
    k_scan3<<<sgrid, 256, 0, stream>>>(dt_b, u_b, xdbl_b, A_log_b, D_b, h0_b, siluz, yb, 1);
    // 9. out_proj (WMMA) + residual, NCHW output
    k_outproj<<<(BL / 16) * 4 / 4, 128, 0, stream>>>(yb, out_proj_w, in, out);
}
